// DownLayer_28604482192055
// MI455X (gfx1250) — compile-verified, matrix operand order checked by reading, not run on hardware
//
#include <hip/hip_runtime.h>
#include <hip/hip_bf16.h>
#include <math.h>

// ---------------------------------------------------------------------------
// Problem constants (fixed by the reference)
// ---------------------------------------------------------------------------
#define BB    8
#define NTOK  3136
#define CIN   128
#define COUT  256
#define N0T   12544
#define HH    56
#define WW    56
#define HO    28
#define WO    28
#define NSAMP 784          // ceil(3136*0.25)
#define KCONV 1152         // 9*128
#define MT_GRAM   (NTOK/16)    // 196
#define CT_ASSIGN (NSAMP/16)   // 49

typedef _Float16 v16h __attribute__((ext_vector_type(16)));
typedef float    v8f  __attribute__((ext_vector_type(8)));

// ---------------------------------------------------------------------------
// WMMA helpers (CDNA5 wave32, 16x16x32 f16 -> f32)
// ---------------------------------------------------------------------------
static __device__ inline v8f wmma_f16(v16h a, v16h b, v8f c) {
  return __builtin_amdgcn_wmma_f32_16x16x32_f16(
      /*neg_a=*/false, a, /*neg_b=*/false, b,
      /*c_mod=*/(short)0, c, /*reuse_a=*/false, /*reuse_b=*/false);
}

// Load 16 f16 operand elements for one lane of a 16x32 A (or 32x16 B) tile.
// Lane l: row/col = l&15, hi = l>>4.  k(j<8)=kb+hi*8+j ; k(j>=8)=kb+16+hi*8+(j-8)
static __device__ inline v16h ld16h(const _Float16* p, int kb, int hi) {
  v16h r;
  const _Float16* p0 = p + kb + hi * 8;
  const _Float16* p1 = p + kb + 16 + hi * 8;
#pragma unroll
  for (int j = 0; j < 8; ++j) { r[j] = p0[j]; r[j + 8] = p1[j]; }
  return r;
}

static __device__ inline v16h ld16f(const float* p, int kb, int hi) {
  v16h r;
  const float* p0 = p + kb + hi * 8;
  const float* p1 = p + kb + 16 + hi * 8;
#pragma unroll
  for (int j = 0; j < 8; ++j) { r[j] = (_Float16)p0[j]; r[j + 8] = (_Float16)p1[j]; }
  return r;
}

// im2col A-operand chunk for the 3x3 stride-2 pad-1 conv (K = ky*384+kx*128+ci)
static __device__ inline v16h ldA_conv(const float* xmap, int b, int oy, int ox,
                                       int kb, int hi) {
  v16h r;
  int kc[2] = { kb + hi * 8, kb + 16 + hi * 8 };
#pragma unroll
  for (int c = 0; c < 2; ++c) {
    int k0 = kc[c];
    int patch = k0 >> 7;          // 0..8
    int ci = k0 & 127;
    int ky = patch / 3, kx = patch - ky * 3;
    int iy = oy * 2 - 1 + ky, ix = ox * 2 - 1 + kx;
    bool ok = (iy >= 0) && (iy < HH) && (ix >= 0) && (ix < WW);
    const float* s = xmap + (((size_t)b * HH + (ok ? iy : 0)) * WW + (ok ? ix : 0)) * CIN + ci;
#pragma unroll
    for (int j = 0; j < 8; ++j)
      r[c * 8 + j] = ok ? (_Float16)s[j] : (_Float16)0.0f;
  }
  return r;
}

// ---------------------------------------------------------------------------
// token2map helpers
// ---------------------------------------------------------------------------
static __device__ inline int cellIdx(const float* pos, int gid, int b, int Hc, int Wc) {
  float x = pos[2 * gid], y = pos[2 * gid + 1];
  x = fminf(fmaxf(x, -1.f), 1.f);
  y = fminf(fmaxf(y, -1.f), 1.f);
  int px = (int)fminf(fmaxf(rintf(0.5f * (x + 1.f) * (float)Wc - 0.5f), 0.f), (float)(Wc - 1));
  int py = (int)fminf(fmaxf(rintf(0.5f * (y + 1.f) * (float)Hc - 0.5f), 0.f), (float)(Hc - 1));
  return py * Wc + px + b * Hc * Wc;
}

__global__ void k_count(const float* __restrict__ pos, float* __restrict__ counts) {
  int gid = blockIdx.x * blockDim.x + threadIdx.x;
  if (gid >= BB * N0T) return;
  int b = gid / N0T;
  atomicAdd(&counts[cellIdx(pos, gid, b, HH, WW)], 1.0f);
}

__global__ void k_scatter_map(const float* __restrict__ x, const float* __restrict__ pos,
                              const int* __restrict__ idxagg, const float* __restrict__ counts,
                              float* __restrict__ xmap) {
  int gid = blockIdx.x;            // B*N0 tokens
  int c = threadIdx.x;             // 128 channels
  int b = gid / N0T;
  int cell = cellIdx(pos, gid, b, HH, WW);
  float val = 1.0f / (counts[cell] + 1e-6f);
  int src = idxagg[gid];
  float sv = x[((size_t)b * NTOK + src) * CIN + c] * val;
  atomicAdd(&xmap[(size_t)cell * CIN + c], sv);
}

// ---------------------------------------------------------------------------
// weight pre-transpose (K-contiguous f16 for WMMA B operands)
// ---------------------------------------------------------------------------
__global__ void k_wtrans(const float* __restrict__ cw, _Float16* __restrict__ wt16) {
  int k = blockIdx.x;   // 0..1151
  int n = threadIdx.x;  // 0..255
  wt16[(size_t)n * KCONV + k] = (_Float16)cw[(size_t)k * COUT + n];
}

__global__ void k_strans(const float* __restrict__ sw, _Float16* __restrict__ st16) {
  int k = blockIdx.x;   // 0..127
  int n = threadIdx.x;  // 0..255
  st16[(size_t)n * CIN + k] = (_Float16)sw[(size_t)k * COUT + n];
}

// ---------------------------------------------------------------------------
// conv as implicit GEMM: M=6272 (B*28*28), K=1152, N=256. One 16x16 tile/wave.
// ---------------------------------------------------------------------------
__global__ void k_conv_wmma(const float* __restrict__ xmap, const _Float16* __restrict__ wt16,
                            const float* __restrict__ cb, float* __restrict__ ymap) {
  int wid = threadIdx.x >> 5, lane = threadIdx.x & 31;
  int tile = blockIdx.x * 4 + wid;            // 6272 tiles
  int mT = tile >> 4, nT = tile & 15;
  int nl = lane & 15, hi = lane >> 4;
  int rowA = mT * 16 + nl;                    // output pixel for this lane's A row
  int bA = rowA / (HO * WO);
  int rr = rowA % (HO * WO);
  int oy = rr / WO, ox = rr % WO;
  const _Float16* Bp = wt16 + (size_t)(nT * 16 + nl) * KCONV;
  v8f acc;
#pragma unroll
  for (int j = 0; j < 8; ++j) acc[j] = 0.f;
  for (int kb = 0; kb < KCONV; kb += 32) {
    v16h a = ldA_conv(xmap, bA, oy, ox, kb, hi);
    v16h bm = ld16h(Bp, kb, hi);
    acc = wmma_f16(a, bm, acc);
  }
  int n = nT * 16 + nl;
  float bias = cb[n];
#pragma unroll
  for (int v = 0; v < 8; ++v) {
    int p = mT * 16 + v + 8 * hi;
    ymap[(size_t)p * COUT + n] = acc[v] + bias;
  }
}

// ---------------------------------------------------------------------------
// skip GEMM: x(25088x128) @ skip_w -> x_tok(25088x256)
// ---------------------------------------------------------------------------
__global__ void k_skip_wmma(const float* __restrict__ x, const _Float16* __restrict__ st16,
                            float* __restrict__ xtok) {
  int wid = threadIdx.x >> 5, lane = threadIdx.x & 31;
  int tile = blockIdx.x * 4 + wid;            // 25088 tiles
  int mT = tile >> 4, nT = tile & 15;
  int nl = lane & 15, hi = lane >> 4;
  const float* Ap = x + (size_t)(mT * 16 + nl) * CIN;
  const _Float16* Bp = st16 + (size_t)(nT * 16 + nl) * CIN;
  v8f acc;
#pragma unroll
  for (int j = 0; j < 8; ++j) acc[j] = 0.f;
#pragma unroll
  for (int kb = 0; kb < CIN; kb += 32) {
    v16h a = ld16f(Ap, kb, hi);
    v16h bm = ld16h(Bp, kb, hi);
    acc = wmma_f16(a, bm, acc);
  }
  int n = nT * 16 + nl;
#pragma unroll
  for (int v = 0; v < 8; ++v) {
    int m = mT * 16 + v + 8 * hi;
    xtok[(size_t)m * COUT + n] = acc[v];
  }
}

// ---------------------------------------------------------------------------
// map2token: per-token weight sums, then weighted gather-add onto x_tok
// ---------------------------------------------------------------------------
__global__ void k_allw(const int* __restrict__ idxagg, const float* __restrict__ aw,
                       float* __restrict__ allw) {
  int gid = blockIdx.x * blockDim.x + threadIdx.x;
  if (gid >= BB * N0T) return;
  int b = gid / N0T;
  atomicAdd(&allw[b * NTOK + idxagg[gid]], aw[gid]);
}

__global__ void k_map2tok(const float* __restrict__ ymap, const float* __restrict__ pos,
                          const int* __restrict__ idxagg, const float* __restrict__ aw,
                          const float* __restrict__ allw, float* __restrict__ xtok) {
  int gid = blockIdx.x;            // B*N0
  int c = threadIdx.x;             // 256
  int b = gid / N0T;
  int cell = cellIdx(pos, gid, b, HO, WO);
  int tokd = b * NTOK + idxagg[gid];
  float val = aw[gid] / (allw[tokd] + 1e-6f);
  atomicAdd(&xtok[(size_t)tokd * COUT + c], ymap[(size_t)cell * COUT + c] * val);
}

// ---------------------------------------------------------------------------
// LayerNorm + conf head; emits f32 (in place), f16 copy, |x|^2 and exp(conf)
// ---------------------------------------------------------------------------
__global__ void k_ln(float* __restrict__ xtok, _Float16* __restrict__ xt16,
                     const float* __restrict__ g, const float* __restrict__ bta,
                     const float* __restrict__ cw, const float* __restrict__ cbv,
                     float* __restrict__ d2, float* __restrict__ wgt) {
  __shared__ float s1[256], s2[256];
  int tok = blockIdx.x, c = threadIdx.x;
  float v = xtok[(size_t)tok * COUT + c];
  s1[c] = v; s2[c] = v * v;
  __syncthreads();
  for (int st = 128; st > 0; st >>= 1) {
    if (c < st) { s1[c] += s1[c + st]; s2[c] += s2[c + st]; }
    __syncthreads();
  }
  float mu = s1[0] * (1.f / 256.f);
  float var = s2[0] * (1.f / 256.f) - mu * mu;
  float y = (v - mu) * rsqrtf(var + 1e-5f) * g[c] + bta[c];
  xtok[(size_t)tok * COUT + c] = y;
  xt16[(size_t)tok * COUT + c] = (_Float16)y;
  __syncthreads();
  s1[c] = y * y; s2[c] = y * cw[c];
  __syncthreads();
  for (int st = 128; st > 0; st >>= 1) {
    if (c < st) { s1[c] += s1[c + st]; s2[c] += s2[c + st]; }
    __syncthreads();
  }
  if (c == 0) { d2[tok] = s1[0]; wgt[tok] = expf(s2[0] + cbv[0]); }
}

// ---------------------------------------------------------------------------
// Pass 1: Gram tiles via WMMA -> per-row top-3 distances -> density; global max
// A row-block (16x256 f16 = 64 VGPR) is register-resident across all 196 j-tiles.
// Next B row-block is software-prefetched (global_prefetch_b8).
// Grid: 392 blocks x 128 threads (exactly 8*196 waves, uniform trip counts)
// ---------------------------------------------------------------------------
__global__ void k_gram_density(const _Float16* __restrict__ xt16, const float* __restrict__ d2,
                               float* __restrict__ den, float* __restrict__ dmax) {
  __shared__ float tile[4][16][16];
  int wid = threadIdx.x >> 5, lane = threadIdx.x & 31;
  int gw = blockIdx.x * 4 + wid;
  int b = gw / MT_GRAM, it = gw % MT_GRAM;
  int nl = lane & 15, hi = lane >> 4;
  const _Float16* Ab = xt16 + ((size_t)b * NTOK + it * 16 + nl) * COUT;
  const float* d2b = d2 + b * NTOK;

  v16h areg[8];
#pragma unroll
  for (int kk = 0; kk < 8; ++kk) areg[kk] = ld16h(Ab, kk * 32, hi);

  float t0 = 3.4e38f, t1 = 3.4e38f, t2 = 3.4e38f, lmax = 0.f;
  for (int jt = 0; jt < MT_GRAM; ++jt) {
    const _Float16* Bp = xt16 + ((size_t)b * NTOK + jt * 16 + nl) * COUT;
    if (jt + 1 < MT_GRAM)
      __builtin_prefetch(xt16 + ((size_t)b * NTOK + (jt + 1) * 16 + nl) * COUT, 0, 0);
    v8f acc;
#pragma unroll
    for (int j = 0; j < 8; ++j) acc[j] = 0.f;
#pragma unroll
    for (int kk = 0; kk < 8; ++kk)
      acc = wmma_f16(areg[kk], ld16h(Bp, kk * 32, hi), acc);
#pragma unroll
    for (int v = 0; v < 8; ++v) {
      int m = v + 8 * hi;
      float dd = d2b[it * 16 + m] + d2b[jt * 16 + nl] - 2.f * acc[v];
      float d = sqrtf(fmaxf(dd, 0.f)) * 0.0625f;    // /sqrt(256)
      tile[wid][m][nl] = d;
      lmax = fmaxf(lmax, d);
    }
    __syncthreads();
    if (hi == 0) {
#pragma unroll
      for (int n = 0; n < 16; ++n) {
        float d = tile[wid][nl][n];
        if (d < t2) {
          if (d < t0) { t2 = t1; t1 = t0; t0 = d; }
          else if (d < t1) { t2 = t1; t1 = d; }
          else t2 = d;
        }
      }
    }
    __syncthreads();
  }
  atomicMax((int*)(dmax + b), __float_as_int(lmax));   // dists >= 0
  if (hi == 0) {
    int row = it * 16 + nl;
    float dens = expf(-(t0 * t0 + t1 * t1 + t2 * t2) * (1.f / 3.f));
    unsigned h = ((unsigned)row * 2654435761u) ^ ((unsigned)b * 0x9E3779B9u);
    dens += (float)(h & 0xFFFFFFu) * 5.9604645e-8f * 1e-6f;  // deterministic tie-break
    den[b * NTOK + row] = dens;
  }
}

// ---------------------------------------------------------------------------
// Pass 2: parent distance (min dist to any higher-density point) -> score
// ---------------------------------------------------------------------------
__global__ void k_parent_score(const _Float16* __restrict__ xt16, const float* __restrict__ d2,
                               const float* __restrict__ den, const float* __restrict__ dmax,
                               float* __restrict__ score) {
  __shared__ float tile[4][16][16];
  __shared__ float colden[4][16];
  int wid = threadIdx.x >> 5, lane = threadIdx.x & 31;
  int gw = blockIdx.x * 4 + wid;
  int b = gw / MT_GRAM, it = gw % MT_GRAM;
  int nl = lane & 15, hi = lane >> 4;
  const _Float16* Ab = xt16 + ((size_t)b * NTOK + it * 16 + nl) * COUT;
  const float* d2b = d2 + b * NTOK;

  v16h areg[8];
#pragma unroll
  for (int kk = 0; kk < 8; ++kk) areg[kk] = ld16h(Ab, kk * 32, hi);

  float myden = den[b * NTOK + it * 16 + nl];   // meaningful for hi==0 lanes
  float pmin = dmax[b];
  for (int jt = 0; jt < MT_GRAM; ++jt) {
    const _Float16* Bp = xt16 + ((size_t)b * NTOK + jt * 16 + nl) * COUT;
    if (jt + 1 < MT_GRAM)
      __builtin_prefetch(xt16 + ((size_t)b * NTOK + (jt + 1) * 16 + nl) * COUT, 0, 0);
    v8f acc;
#pragma unroll
    for (int j = 0; j < 8; ++j) acc[j] = 0.f;
#pragma unroll
    for (int kk = 0; kk < 8; ++kk)
      acc = wmma_f16(areg[kk], ld16h(Bp, kk * 32, hi), acc);
#pragma unroll
    for (int v = 0; v < 8; ++v) {
      int m = v + 8 * hi;
      float dd = d2b[it * 16 + m] + d2b[jt * 16 + nl] - 2.f * acc[v];
      tile[wid][m][nl] = sqrtf(fmaxf(dd, 0.f)) * 0.0625f;
    }
    if (hi == 0) colden[wid][nl] = den[b * NTOK + jt * 16 + nl];
    __syncthreads();
    if (hi == 0) {
#pragma unroll
      for (int n = 0; n < 16; ++n)
        if (colden[wid][n] > myden) pmin = fminf(pmin, tile[wid][nl][n]);
    }
    __syncthreads();
  }
  if (hi == 0) score[b * NTOK + it * 16 + nl] = pmin * myden;
}

// ---------------------------------------------------------------------------
// per-batch descending bitonic top-784 (stable: ties -> smaller index first)
// ---------------------------------------------------------------------------
__global__ void __launch_bounds__(1024) k_sort_topk(const float* __restrict__ score,
                                                    int* __restrict__ idxdown) {
  __shared__ float sv[4096];
  __shared__ int si[4096];
  int b = blockIdx.x, tid = threadIdx.x;
  for (int i = tid; i < 4096; i += 1024) {
    if (i < NTOK) { sv[i] = score[b * NTOK + i]; si[i] = i; }
    else          { sv[i] = -3.4e38f;            si[i] = 0x7fffffff; }
  }
  __syncthreads();
  for (int k = 2; k <= 4096; k <<= 1) {
    for (int j = k >> 1; j > 0; j >>= 1) {
      for (int i = tid; i < 4096; i += 1024) {
        int l = i ^ j;
        if (l > i) {
          float va = sv[i], vb = sv[l];
          int ia = si[i], ib = si[l];
          bool before = (va > vb) || (va == vb && ia < ib);
          bool dir = ((i & k) == 0);
          if (dir ? !before : before) { sv[i] = vb; sv[l] = va; si[i] = ib; si[l] = ia; }
        }
      }
      __syncthreads();
    }
  }
  for (int i = tid; i < NSAMP; i += 1024) idxdown[b * NSAMP + i] = si[i];
}

__global__ void k_d2c(const int* __restrict__ idxdown, const float* __restrict__ d2,
                      float* __restrict__ d2c) {
  int gid = blockIdx.x * blockDim.x + threadIdx.x;
  if (gid >= BB * NSAMP) return;
  int b = gid / NSAMP;
  d2c[gid] = d2[b * NTOK + idxdown[gid]];
}

// ---------------------------------------------------------------------------
// center assignment: argmin over 784 centers per token, via WMMA tiles.
// B (token column block) is loop-invariant -> register-resident; A (centers)
// gathered per c-tile through idxdown.
// ---------------------------------------------------------------------------
__global__ void k_assign(const _Float16* __restrict__ xt16, const float* __restrict__ d2,
                         const float* __restrict__ d2c, const int* __restrict__ idxdown,
                         int* __restrict__ idxcl) {
  __shared__ float smv[4][32];
  __shared__ int smc[4][32];
  int wid = threadIdx.x >> 5, lane = threadIdx.x & 31;
  int gw = blockIdx.x * 4 + wid;
  int b = gw / MT_GRAM, it = gw % MT_GRAM;
  int nl = lane & 15, hi = lane >> 4;
  int tok = it * 16 + nl;
  const _Float16* Bp = xt16 + ((size_t)b * NTOK + tok) * COUT;

  v16h breg[8];
#pragma unroll
  for (int kk = 0; kk < 8; ++kk) breg[kk] = ld16h(Bp, kk * 32, hi);

  float d2t = d2[b * NTOK + tok];
  float minv = 3.4e38f;
  int minc = 0x7fffffff;
  for (int ct = 0; ct < CT_ASSIGN; ++ct) {
    int crow = ct * 16 + nl;
    int ctok = idxdown[b * NSAMP + crow];
    const _Float16* Ab = xt16 + ((size_t)b * NTOK + ctok) * COUT;
    v8f acc;
#pragma unroll
    for (int j = 0; j < 8; ++j) acc[j] = 0.f;
#pragma unroll
    for (int kk = 0; kk < 8; ++kk)
      acc = wmma_f16(ld16h(Ab, kk * 32, hi), breg[kk], acc);
#pragma unroll
    for (int v = 0; v < 8; ++v) {
      int c = ct * 16 + v + 8 * hi;
      float dd = d2c[b * NSAMP + c] + d2t - 2.f * acc[v];
      float d = sqrtf(fmaxf(dd, 0.f)) * 0.0625f;
      if (d < minv || (d == minv && c < minc)) { minv = d; minc = c; }
    }
  }
  smv[wid][lane] = minv; smc[wid][lane] = minc;
  __syncthreads();
  if (hi == 0) {
    float v2 = smv[wid][lane + 16];
    int c2 = smc[wid][lane + 16];
    if (v2 < minv || (v2 == minv && c2 < minc)) { minv = v2; minc = c2; }
    idxcl[b * NTOK + tok] = minc;
  }
}

__global__ void k_override(const int* __restrict__ idxdown, int* __restrict__ idxcl) {
  int gid = blockIdx.x * blockDim.x + threadIdx.x;
  if (gid >= BB * NSAMP) return;
  int b = gid / NSAMP, c = gid % NSAMP;
  idxcl[b * NTOK + idxdown[gid]] = c;
}

// ---------------------------------------------------------------------------
// weighted merge into clusters + final outputs
// ---------------------------------------------------------------------------
__global__ void k_allw2(const int* __restrict__ idxcl, const float* __restrict__ wgt,
                        float* __restrict__ allw2) {
  int gid = blockIdx.x * blockDim.x + threadIdx.x;
  if (gid >= BB * NTOK) return;
  int b = gid / NTOK;
  atomicAdd(&allw2[b * NSAMP + idxcl[gid]], wgt[gid]);
}

__global__ void k_normw(const int* __restrict__ idxcl, const float* __restrict__ wgt,
                        const float* __restrict__ allw2, float* __restrict__ normw) {
  int gid = blockIdx.x * blockDim.x + threadIdx.x;
  if (gid >= BB * NTOK) return;
  int b = gid / NTOK;
  normw[gid] = wgt[gid] / (allw2[b * NSAMP + idxcl[gid]] + 1e-6f);
}

__global__ void k_xdown(const float* __restrict__ xtok, const int* __restrict__ idxcl,
                        const float* __restrict__ normw, float* __restrict__ xdown) {
  int gid = blockIdx.x;  // B*NTOK
  int c = threadIdx.x;   // 256
  int b = gid / NTOK;
  int cl = idxcl[gid];
  atomicAdd(&xdown[((size_t)b * NSAMP + cl) * COUT + c],
            xtok[(size_t)gid * COUT + c] * normw[gid]);
}

__global__ void k_final1(const int* __restrict__ idxagg, const int* __restrict__ idxcl,
                         const float* __restrict__ normw, const float* __restrict__ aw,
                         float* __restrict__ outIdx, float* __restrict__ outAw,
                         float* __restrict__ maxw) {
  int gid = blockIdx.x * blockDim.x + threadIdx.x;
  if (gid >= BB * N0T) return;
  int b = gid / N0T;
  int t = idxagg[gid];
  int c = idxcl[b * NTOK + t];
  outIdx[gid] = (float)c;
  float v = aw[gid] * normw[b * NTOK + t];
  outAw[gid] = v;
  atomicMax((int*)(maxw + b), __float_as_int(v));   // v > 0
}

__global__ void k_final2(float* __restrict__ outAw, const float* __restrict__ maxw) {
  int gid = blockIdx.x * blockDim.x + threadIdx.x;
  if (gid >= BB * N0T) return;
  int b = gid / N0T;
  outAw[gid] /= maxw[b];
}

// ---------------------------------------------------------------------------
// host launcher
// ---------------------------------------------------------------------------
extern "C" void kernel_launch(void* const* d_in, const int* in_sizes, int n_in,
                              void* d_out, int out_size, void* d_ws, size_t ws_size,
                              hipStream_t stream) {
  const float* x       = (const float*)d_in[0];
  const float* pos     = (const float*)d_in[1];
  const int*   idxagg  = (const int*)d_in[2];
  const float* aw      = (const float*)d_in[3];
  // d_in[4..6] = H, W, N_grid scalars on device: fixed constants, unused here
  const float* conv_w  = (const float*)d_in[7];
  const float* conv_b  = (const float*)d_in[8];
  const float* skip_w  = (const float*)d_in[9];
  const float* ln_g    = (const float*)d_in[10];
  const float* ln_b    = (const float*)d_in[11];
  const float* conf_w  = (const float*)d_in[12];
  const float* conf_b  = (const float*)d_in[13];

  // workspace layout (256-byte aligned)
  size_t off = 0;
  auto alloc = [&](size_t bytes) -> void* {
    off = (off + 255) & ~(size_t)255;
    void* p = (char*)d_ws + off;
    off += bytes;
    return p;
  };
  float*    counts = (float*)alloc((size_t)BB * HH * WW * 4);
  float*    xmap   = (float*)alloc((size_t)BB * HH * WW * CIN * 4);
  float*    ymap   = (float*)alloc((size_t)BB * HO * WO * COUT * 4);
  float*    xtok   = (float*)alloc((size_t)BB * NTOK * COUT * 4);
  _Float16* xt16   = (_Float16*)alloc((size_t)BB * NTOK * COUT * 2);
  _Float16* wt16   = (_Float16*)alloc((size_t)COUT * KCONV * 2);
  _Float16* st16   = (_Float16*)alloc((size_t)COUT * CIN * 2);
  float*    allw   = (float*)alloc((size_t)BB * NTOK * 4);
  float*    d2     = (float*)alloc((size_t)BB * NTOK * 4);
  float*    wgt    = (float*)alloc((size_t)BB * NTOK * 4);
  float*    den    = (float*)alloc((size_t)BB * NTOK * 4);
  float*    score  = (float*)alloc((size_t)BB * NTOK * 4);
  float*    dmax   = (float*)alloc((size_t)BB * 4);
  int*      idxdn  = (int*)alloc((size_t)BB * NSAMP * 4);
  float*    d2c    = (float*)alloc((size_t)BB * NSAMP * 4);
  int*      idxcl  = (int*)alloc((size_t)BB * NTOK * 4);
  float*    allw2  = (float*)alloc((size_t)BB * NSAMP * 4);
  float*    normw  = (float*)alloc((size_t)BB * NTOK * 4);
  float*    maxw   = (float*)alloc((size_t)BB * 4);

  float* xdown  = (float*)d_out;                                        // B*784*256
  float* outIdx = xdown + (size_t)BB * NSAMP * COUT;                    // B*12544
  float* outAw  = outIdx + (size_t)BB * N0T;                            // B*12544

  // zero accumulators (stream ops: graph-capture safe)
  hipMemsetAsync(counts, 0, (size_t)BB * HH * WW * 4, stream);
  hipMemsetAsync(xmap,   0, (size_t)BB * HH * WW * CIN * 4, stream);
  hipMemsetAsync(allw,   0, (size_t)BB * NTOK * 4, stream);
  hipMemsetAsync(dmax,   0, (size_t)BB * 4, stream);
  hipMemsetAsync(allw2,  0, (size_t)BB * NSAMP * 4, stream);
  hipMemsetAsync(maxw,   0, (size_t)BB * 4, stream);
  hipMemsetAsync(xdown,  0, (size_t)BB * NSAMP * COUT * 4, stream);

  const int NTOT0 = BB * N0T;   // 100352
  const int NTOTT = BB * NTOK;  // 25088

  // stage A: token2map
  k_count<<<(NTOT0 + 255) / 256, 256, 0, stream>>>(pos, counts);
  k_scatter_map<<<NTOT0, CIN, 0, stream>>>(x, pos, idxagg, counts, xmap);

  // weight pre-transpose to f16
  k_wtrans<<<KCONV, COUT, 0, stream>>>(conv_w, wt16);
  k_strans<<<CIN, COUT, 0, stream>>>(skip_w, st16);

  // stage B: conv (WMMA) and skip GEMM (WMMA)
  k_conv_wmma<<<(BB * HO * WO / 16) * 16 / 4, 128, 0, stream>>>(xmap, wt16, conv_b, ymap); // 1568 blocks
  k_skip_wmma<<<(NTOTT / 16) * 16 / 4, 128, 0, stream>>>(x, st16, xtok);                   // 6272 blocks

  // stage C: map2token accumulate onto skip result
  k_allw<<<(NTOT0 + 255) / 256, 256, 0, stream>>>(idxagg, aw, allw);
  k_map2tok<<<NTOT0, COUT, 0, stream>>>(ymap, pos, idxagg, aw, allw, xtok);

  // stage D: LayerNorm + conf
  k_ln<<<NTOTT, COUT, 0, stream>>>(xtok, xt16, ln_g, ln_b, conf_w, conf_b, d2, wgt);

  // stage E: DPC-KNN clustering (WMMA Gram tiles, recomputed per pass)
  k_gram_density<<<BB * MT_GRAM / 4, 128, 0, stream>>>(xt16, d2, den, dmax);     // 392 blocks
  k_parent_score<<<BB * MT_GRAM / 4, 128, 0, stream>>>(xt16, d2, den, dmax, score);
  k_sort_topk<<<BB, 1024, 0, stream>>>(score, idxdn);
  k_d2c<<<(BB * NSAMP + 255) / 256, 256, 0, stream>>>(idxdn, d2, d2c);
  k_assign<<<BB * MT_GRAM / 4, 128, 0, stream>>>(xt16, d2, d2c, idxdn, idxcl);
  k_override<<<(BB * NSAMP + 255) / 256, 256, 0, stream>>>(idxdn, idxcl);

  // stage F: weighted merge + outputs
  k_allw2<<<(NTOTT + 255) / 256, 256, 0, stream>>>(idxcl, wgt, allw2);
  k_normw<<<(NTOTT + 255) / 256, 256, 0, stream>>>(idxcl, wgt, allw2, normw);
  k_xdown<<<NTOTT, COUT, 0, stream>>>(xtok, idxcl, normw, xdown);
  k_final1<<<(NTOT0 + 255) / 256, 256, 0, stream>>>(idxagg, idxcl, normw, aw, outIdx, outAw, maxw);
  k_final2<<<(NTOT0 + 255) / 256, 256, 0, stream>>>(outAw, maxw);
}
